// MGTSketchEncoder_31997506355781
// MI455X (gfx1250) — matrix operand order repaired and checked by reading, not compiled, not run on hardware
//
#include <hip/hip_runtime.h>
#include <hip/hip_bf16.h>

typedef __attribute__((ext_vector_type(16))) _Float16 v16h;
typedef __attribute__((ext_vector_type(8)))  _Float16 v8h;
typedef __attribute__((ext_vector_type(8)))  float    v8f;

#define DEV __device__ __forceinline__

// Problem constants
constexpr int Bb = 32, Ss = 256, Ee = 256, Hh = 8, Ll = 4, DKk = 32;
constexpr int Mrows = Bb * Ss; // 8192

// ---------------- WMMA fragment helpers (gfx1250, wave32) ----------------
// A-matrix 16x32 f16 (ISA 7.12.2): lane m=lane&15, half=lane>>4;
//   a[0..7]  = A[m, k0 + half*8 + j]
//   a[8..15] = A[m, k0 + 16 + half*8 + j]
DEV v16h load_fragA(const _Float16* row, int k0, int half) {
  v8h x0 = *(const v8h*)(row + k0 + half * 8);
  v8h x1 = *(const v8h*)(row + k0 + 16 + half * 8);
  return __builtin_shufflevector(x0, x1, 0,1,2,3,4,5,6,7,8,9,10,11,12,13,14,15);
}
// B-matrix 32x16 f16 (column-major storage Bt[N][K]): lane n=lane&15, half=lane>>4;
//   b[0..7]  = Bt[n, k0 + half*8 + j]
//   b[8..15] = Bt[n, k0 + 16 + half*8 + j]
DEV v16h load_fragB(const _Float16* row, int k0, int half) {
  v8h x0 = *(const v8h*)(row + k0 + half * 8);
  v8h x1 = *(const v8h*)(row + k0 + 16 + half * 8);
  return __builtin_shufflevector(x0, x1, 0,1,2,3,4,5,6,7,8,9,10,11,12,13,14,15);
}
DEV v8f wmma16(v16h a, v16h b, v8f c) {
  return __builtin_amdgcn_wmma_f32_16x16x32_f16(false, a, false, b, (short)0, c, false, false);
}
DEV v8f zero8() { v8f z;
#pragma unroll
  for (int i = 0; i < 8; ++i) z[i] = 0.0f; return z; }

// ---------------- flags + lengths ----------------
__global__ __launch_bounds__(256) void k_flags(const float* __restrict__ data,
                                               const float* __restrict__ mask,
                                               int* __restrict__ flags,
                                               int* __restrict__ lens) {
  int b = blockIdx.x, s = threadIdx.x;
  float mv = mask[b * Ss + s];
  const float* p = data + ((size_t)b * Ss + s) * 5;
  int f = 0;
  if (p[2] == 1.0f) f = 100;
  if (p[3] == 1.0f || p[4] == 1.0f) f = 101;
  if (mv == 0.0f) f = 102;
  flags[b * Ss + s] = f;
  __shared__ int cnt[256];
  cnt[s] = (mv != 0.0f) ? 1 : 0;
  __syncthreads();
  for (int st = 128; st > 0; st >>= 1) { if (s < st) cnt[s] += cnt[s + st]; __syncthreads(); }
  if (s == 0) lens[b] = cnt[0];
}

// ---------------- embedding ----------------
__global__ __launch_bounds__(256) void k_embed(const float* __restrict__ data,
                                               const int* __restrict__ flags,
                                               const float* __restrict__ Wc,
                                               const float* __restrict__ bc,
                                               const float* __restrict__ ftab,
                                               const float* __restrict__ ptab,
                                               float* __restrict__ x,
                                               _Float16* __restrict__ xh) {
  int bs = blockIdx.x, e = threadIdx.x;
  int s = bs & (Ss - 1);
  float c0 = data[(size_t)bs * 5];
  float c1 = data[(size_t)bs * 5 + 1];
  int f = flags[bs];
  float v = c0 * Wc[e] + c1 * Wc[Ee + e] + bc[e] + ftab[(size_t)f * Ee + e] + ptab[(size_t)s * Ee + e];
  x[(size_t)bs * Ee + e] = v;
  xh[(size_t)bs * Ee + e] = (_Float16)v;
}

// ---------------- generic batched transpose + f32->f16 cast ----------------
// dst[mat][n][k] = src[mat][k][n]  (dst row-major [N,K] per matrix)
__global__ __launch_bounds__(256) void k_transpose(const float* __restrict__ src,
                                                   _Float16* __restrict__ dst,
                                                   int mats, int K, int N) {
  size_t t = (size_t)blockIdx.x * 256 + threadIdx.x;
  size_t total = (size_t)mats * K * N;
  if (t >= total) return;
  size_t per = (size_t)K * N;
  size_t mat = t / per;
  size_t rkn = t % per;
  int n = (int)(rkn / K), k = (int)(rkn % K);
  dst[t] = (_Float16)src[mat * per + (size_t)k * N + n];
}

// ---------------- WMMA GEMM: C = A[M,K] x Bt[N,K]^T, M=8192, N=256 ----------------
// MODE 2: f16 out scattered to Q/K layout [B,H,S,DK]
// MODE 3: f16 out scattered to V^T layout [B,H,DK,S]
// MODE 4: f16 out = res_f32 + acc, into concat buffer (ld=768, +coloff)
// MODE 5: f32 out = acc + bias[n]
// MODE 6: f32 out = res_f32 + relu(acc + bias[n])
template<int MODE>
__global__ __launch_bounds__(256)
void k_gemm(const _Float16* __restrict__ A, const _Float16* __restrict__ Bt, int K,
            float* __restrict__ outF, _Float16* __restrict__ outH,
            const float* __restrict__ res, const float* __restrict__ bias, int coloff) {
  int tid = threadIdx.x;
  int w = tid >> 5, lane = tid & 31;
  int half = lane >> 4, l15 = lane & 15;
  int wr = w & 3, wc = w >> 2;
  int row0 = blockIdx.y * 128 + wr * 32;
  int col0 = blockIdx.x * 128 + wc * 64;

  v8f acc[2][4];
#pragma unroll
  for (int i = 0; i < 2; ++i)
#pragma unroll
    for (int j = 0; j < 4; ++j) acc[i][j] = zero8();

  const _Float16* aRow0 = A + (size_t)(row0 + l15) * K;
  const _Float16* aRow1 = A + (size_t)(row0 + 16 + l15) * K;
  const _Float16* bRow[4];
#pragma unroll
  for (int nt = 0; nt < 4; ++nt) bRow[nt] = Bt + (size_t)(col0 + nt * 16 + l15) * K;

  int kIters = K >> 5;
  for (int kt = 0; kt < kIters; ++kt) {
    int k0 = kt << 5;
    __builtin_prefetch(aRow0 + k0 + 64, 0, 1);   // global_prefetch_b8
    v16h a0 = load_fragA(aRow0, k0, half);
    v16h a1 = load_fragA(aRow1, k0, half);
    v16h b0 = load_fragB(bRow[0], k0, half);
    v16h b1 = load_fragB(bRow[1], k0, half);
    v16h b2 = load_fragB(bRow[2], k0, half);
    v16h b3 = load_fragB(bRow[3], k0, half);
    acc[0][0] = wmma16(a0, b0, acc[0][0]);
    acc[0][1] = wmma16(a0, b1, acc[0][1]);
    acc[0][2] = wmma16(a0, b2, acc[0][2]);
    acc[0][3] = wmma16(a0, b3, acc[0][3]);
    acc[1][0] = wmma16(a1, b0, acc[1][0]);
    acc[1][1] = wmma16(a1, b1, acc[1][1]);
    acc[1][2] = wmma16(a1, b2, acc[1][2]);
    acc[1][3] = wmma16(a1, b3, acc[1][3]);
  }

  // C/D layout: c[r] = D[r + 8*half, lane&15] (ISA 7.12.2)
#pragma unroll
  for (int mt = 0; mt < 2; ++mt)
#pragma unroll
    for (int nt = 0; nt < 4; ++nt)
#pragma unroll
      for (int r = 0; r < 8; ++r) {
        int m = row0 + mt * 16 + r + 8 * half;
        int n = col0 + nt * 16 + l15;
        float v = acc[mt][nt][r];
        if (MODE == 2) {
          int b = m >> 8, s = m & 255, hh = n >> 5, dk = n & 31;
          outH[(((size_t)(b * Hh + hh) * Ss) + s) * DKk + dk] = (_Float16)v;
        } else if (MODE == 3) {
          int b = m >> 8, s = m & 255, hh = n >> 5, dk = n & 31;
          outH[(((size_t)(b * Hh + hh) * DKk) + dk) * Ss + s] = (_Float16)v;
        } else if (MODE == 4) {
          outH[(size_t)m * 768 + coloff + n] = (_Float16)(res[(size_t)m * Ee + n] + v);
        } else if (MODE == 5) {
          outF[(size_t)m * Ee + n] = v + bias[n];
        } else if (MODE == 6) {
          float t = v + bias[n];
          outF[(size_t)m * Ee + n] = res[(size_t)m * Ee + n] + (t > 0.0f ? t : 0.0f);
        }
      }
}

// ---------------- attention: per (b,h) logits -> softmax -> P*V ----------------
// Templated on group G so the adjacency structure is compile-time.
// grid (4, B*H), block 128 (4 waves, 16 query rows each)
template<int G>
__global__ __launch_bounds__(128)
void k_attn(const _Float16* __restrict__ qb, const _Float16* __restrict__ kb,
            const _Float16* __restrict__ vt, const int* __restrict__ flags,
            const int* __restrict__ lens, _Float16* __restrict__ headb) {
  __shared__ int sF[Ss + 2];            // +2 zero sentinels: sF[q+1] always safe
  __shared__ _Float16 sP[4][16][272];   // padded rows to spread LDS banks

  int tid = threadIdx.x;
  int w = tid >> 5, lane = tid & 31;
  int half = lane >> 4, l15 = lane & 15;
  int bh = blockIdx.y;
  int b = bh >> 3, h = bh & 7;
  int len = lens[b];
  for (int i = tid; i < Ss; i += 128) sF[i] = flags[b * Ss + i];
  if (tid < 2) sF[Ss + tid] = 0;
  __syncthreads();

  int q0 = blockIdx.x * 64 + w * 16;
  const size_t bhS = (size_t)(b * Hh + h) * Ss;
  const int lenm1 = len - 1, lenm2 = len - 2;
  const int lgt1 = (len > 1) ? 1 : 0;

  // Q fragment: rows q0..q0+15, K = DK = 32 (one fragment)
  const _Float16* qrow = qb + (bhS + q0 + l15) * DKk;
  v16h aq = load_fragA(qrow, 0, half);

  // Per-row (q) adjacency predicates, branchless 0/1 ints
  int qv[8], dq[8], aq1[8], aq2[8], wq[8];
#pragma unroll
  for (int r = 0; r < 8; ++r) {
    int q = q0 + r + 8 * half;
    int fq = sF[q], fq1 = sF[q + 1];
    qv[r] = q;
    dq[r] = (q < len) ? 1 : 0;
    if (G == 2) {
      aq1[r] = ((fq == 101) ? 1 : 0) & ((q < lenm1) ? 1 : 0);
      aq2[r] = 0;
      wq[r]  = ((q == 0) ? 1 : 0);                 // wrap: q==0 with kk==len-1
    } else {
      aq1[r] = ((fq == 100) ? 1 : 0) & ((q < lenm1) ? 1 : 0);
      aq2[r] = (G == 1) ? (((fq == 100) ? 1 : 0) & ((fq1 == 100) ? 1 : 0) & ((q < lenm2) ? 1 : 0)) : 0;
      wq[r]  = 0;
    }
  }

  const float scale = 0.17677669529663687f; // 1/sqrt(32)
  float lgt[16][8];

  // logits = Q K^T * scale + bias   (K stored [S,DK] == Bt layout for free)
#pragma unroll
  for (int j = 0; j < 16; ++j) {
    int kk = j * 16 + l15;
    int fk = sF[kk], fk1 = sF[kk + 1];
    int ak1, ak2, wk;
    if (G == 2) {
      ak1 = ((fk == 101) ? 1 : 0) & ((kk < lenm1) ? 1 : 0);
      ak2 = 0;
      wk  = ((kk == lenm1) ? 1 : 0) & lgt1;        // wrap partner
    } else {
      ak1 = ((fk == 100) ? 1 : 0) & ((kk < lenm1) ? 1 : 0);
      ak2 = (G == 1) ? (((fk == 100) ? 1 : 0) & ((fk1 == 100) ? 1 : 0) & ((kk < lenm2) ? 1 : 0)) : 0;
      wk  = 0;
    }
    const _Float16* krow = kb + (bhS + kk) * DKk;
    v16h bk = load_fragB(krow, 0, half);
    v8f c = zero8();
    c = wmma16(aq, bk, c);
#pragma unroll
    for (int r = 0; r < 8; ++r) {
      int q = qv[r];
      int d = kk - q;
      int adj = (((d == 0) ? 1 : 0) & dq[r])
              | (((d == 1) ? 1 : 0) & aq1[r])
              | (((d == -1) ? 1 : 0) & ak1);
      if (G == 1) {
        adj |= (((d == 2) ? 1 : 0) & aq2[r])
             | (((d == -2) ? 1 : 0) & ak2);
      }
      if (G == 2) {
        adj |= (wq[r] & wk)                                           // q==0, kk==len-1
             | (((kk == 0) ? 1 : 0) & ((q == lenm1) ? 1 : 0) & lgt1); // kk==0, q==len-1
      }
      lgt[j][r] = c[r] * scale + (adj ? 0.0f : -1.0e10f);
    }
  }

  // row softmax: reduce in-lane (16 tiles) then across the 16-lane half-group
#pragma unroll
  for (int r = 0; r < 8; ++r) {
    float m = -3.4e38f;
#pragma unroll
    for (int j = 0; j < 16; ++j) m = fmaxf(m, lgt[j][r]);
#pragma unroll
    for (int off = 1; off < 16; off <<= 1) m = fmaxf(m, __shfl_xor(m, off, 32));
    float s = 0.0f;
#pragma unroll
    for (int j = 0; j < 16; ++j) { float e = __expf(lgt[j][r] - m); lgt[j][r] = e; s += e; }
#pragma unroll
    for (int off = 1; off < 16; off <<= 1) s += __shfl_xor(s, off, 32);
    float inv = 1.0f / s;
    int rl = r + 8 * half;
#pragma unroll
    for (int j = 0; j < 16; ++j) sP[w][rl][j * 16 + l15] = (_Float16)(lgt[j][r] * inv);
  }
  __syncthreads();

  // O = P[16,256] * V  (V^T stored [DK,S] == Bt layout)
  v8f acc0 = zero8(), acc1 = zero8();
  const _Float16* prow = &sP[w][l15][0];
  const _Float16* vrow0 = vt + ((size_t)(b * Hh + h) * DKk + l15) * Ss;
  const _Float16* vrow1 = vt + ((size_t)(b * Hh + h) * DKk + 16 + l15) * Ss;
#pragma unroll
  for (int c = 0; c < 8; ++c) {
    int k0 = c * 32;
    v16h ap = load_fragA(prow, k0, half);
    v16h b0 = load_fragB(vrow0, k0, half);
    v16h b1 = load_fragB(vrow1, k0, half);
    acc0 = wmma16(ap, b0, acc0);
    acc1 = wmma16(ap, b1, acc1);
  }
#pragma unroll
  for (int r = 0; r < 8; ++r) {
    int q = q0 + r + 8 * half;
    size_t base = ((size_t)b * Ss + q) * Ee + h * DKk;
    headb[base + l15] = (_Float16)acc0[r];
    headb[base + 16 + l15] = (_Float16)acc1[r];
  }
}

// ---------------- batchnorm stats (population mean/var per channel) ----------------
__global__ __launch_bounds__(256) void k_bnstats(const float* __restrict__ h, float* __restrict__ mv) {
  int e = blockIdx.x, t = threadIdx.x;
  float s = 0.0f, s2 = 0.0f;
  for (int r = t; r < Mrows; r += 256) { float v = h[(size_t)r * Ee + e]; s += v; s2 += v * v; }
  __shared__ float r1[256], r2[256];
  r1[t] = s; r2[t] = s2; __syncthreads();
  for (int st = 128; st > 0; st >>= 1) {
    if (t < st) { r1[t] += r1[t + st]; r2[t] += r2[t + st]; }
    __syncthreads();
  }
  if (t == 0) { float m = r1[0] / (float)Mrows; mv[e] = m; mv[Ee + e] = r2[0] / (float)Mrows - m * m; }
}

__global__ __launch_bounds__(256) void k_bnapply(const float* __restrict__ in, const float* __restrict__ mv,
                                                 const float* __restrict__ gamma, const float* __restrict__ beta,
                                                 float* __restrict__ outF, _Float16* __restrict__ outH) {
  int e = threadIdx.x;
  size_t row = blockIdx.x;
  float v = in[row * Ee + e];
  float o = gamma[e] * (v - mv[e]) * rsqrtf(mv[Ee + e] + 1e-5f) + beta[e];
  outF[row * Ee + e] = o;
  outH[row * Ee + e] = (_Float16)o;
}

// ---------------- masked mean over sequence ----------------
__global__ __launch_bounds__(256) void k_final(const float* __restrict__ x, const float* __restrict__ mask,
                                               const int* __restrict__ lens, float* __restrict__ out) {
  int b = blockIdx.x, e = threadIdx.x;
  float s = 0.0f;
  for (int ss = 0; ss < Ss; ++ss) s += x[((size_t)b * Ss + ss) * Ee + e] * mask[b * Ss + ss];
  float denom = (float)lens[b];
  if (denom < 1.0f) denom = 1.0f;
  out[b * Ee + e] = s / denom;
}

// ---------------- host orchestration ----------------
extern "C" void kernel_launch(void* const* d_in, const int* in_sizes, int n_in,
                              void* d_out, int out_size, void* d_ws, size_t ws_size,
                              hipStream_t stream) {
  const float* stroke = (const float*)d_in[0];
  const float* mask   = (const float*)d_in[1];
  const float* Wc  = (const float*)d_in[2];
  const float* bc  = (const float*)d_in[3];
  const float* ftab = (const float*)d_in[4];
  const float* ptab = (const float*)d_in[5];
  const float* Wq  = (const float*)d_in[6];
  const float* Wk  = (const float*)d_in[7];
  const float* Wv  = (const float*)d_in[8];
  const float* Wo  = (const float*)d_in[9];
  const float* Wf  = (const float*)d_in[10];
  const float* bfp = (const float*)d_in[11];
  const float* g1  = (const float*)d_in[12];
  const float* b1  = (const float*)d_in[13];
  const float* Wff = (const float*)d_in[14];
  const float* bffp= (const float*)d_in[15];
  const float* g2  = (const float*)d_in[16];
  const float* b2  = (const float*)d_in[17];

  char* ws = (char*)d_ws;
  size_t off = 0;
  auto take = [&](size_t bytes) -> char* {
    char* p = ws + off;
    off = (off + bytes + 255) & ~(size_t)255;
    return p;
  };
  int*      flags = (int*)take((size_t)Mrows * 4);
  int*      lens  = (int*)take((size_t)Bb * 4);
  float*    x     = (float*)take((size_t)Mrows * Ee * 4);
  _Float16* xh    = (_Float16*)take((size_t)Mrows * Ee * 2);
  _Float16* qb    = (_Float16*)take((size_t)Mrows * Ee * 2);
  _Float16* kbuf  = (_Float16*)take((size_t)Mrows * Ee * 2);
  _Float16* vt    = (_Float16*)take((size_t)Mrows * Ee * 2);
  _Float16* headb = (_Float16*)take((size_t)Mrows * Ee * 2);
  _Float16* cat   = (_Float16*)take((size_t)Mrows * 768 * 2);
  float*    hbuf  = (float*)take((size_t)Mrows * Ee * 4);
  float*    hbn   = (float*)take((size_t)Mrows * Ee * 4);
  _Float16* hbn16 = (_Float16*)take((size_t)Mrows * Ee * 2);
  float*    mv    = (float*)take(2 * Ee * 4);
  _Float16* wqT   = (_Float16*)take((size_t)12 * 65536 * 2);
  _Float16* wkT   = (_Float16*)take((size_t)12 * 65536 * 2);
  _Float16* wvT   = (_Float16*)take((size_t)12 * 65536 * 2);
  _Float16* woT   = (_Float16*)take((size_t)12 * 65536 * 2);
  _Float16* wfT   = (_Float16*)take((size_t)4 * 768 * 256 * 2);
  _Float16* wffT  = (_Float16*)take((size_t)4 * 65536 * 2);

  k_flags<<<Bb, 256, 0, stream>>>(stroke, mask, flags, lens);
  k_embed<<<Mrows, 256, 0, stream>>>(stroke, flags, Wc, bc, ftab, ptab, x, xh);

  // Weight prep: f32 -> f16, transposed to Bt[N,K]
  k_transpose<<<(96 * 8192 + 255) / 256, 256, 0, stream>>>(Wq, wqT, 96, 256, 32);
  k_transpose<<<(96 * 8192 + 255) / 256, 256, 0, stream>>>(Wk, wkT, 96, 256, 32);
  k_transpose<<<(96 * 8192 + 255) / 256, 256, 0, stream>>>(Wv, wvT, 96, 256, 32);
  k_transpose<<<(12 * 65536 + 255) / 256, 256, 0, stream>>>(Wo, woT, 12, 256, 256);
  k_transpose<<<(4 * 196608 + 255) / 256, 256, 0, stream>>>(Wf, wfT, 4, 768, 256);
  k_transpose<<<(4 * 65536 + 255) / 256, 256, 0, stream>>>(Wff, wffT, 4, 256, 256);

  dim3 gg(2, 64);
  dim3 ga(4, Bb * Hh);
  for (int l = 0; l < Ll; ++l) {
    for (int g = 0; g < 3; ++g) {
      size_t lg = (size_t)(l * 3 + g) * 65536;
      k_gemm<2><<<gg, 256, 0, stream>>>(xh, wqT + lg, 256, nullptr, qb, nullptr, nullptr, 0);
      k_gemm<2><<<gg, 256, 0, stream>>>(xh, wkT + lg, 256, nullptr, kbuf, nullptr, nullptr, 0);
      k_gemm<3><<<gg, 256, 0, stream>>>(xh, wvT + lg, 256, nullptr, vt, nullptr, nullptr, 0);
      if (g == 0)      k_attn<0><<<ga, 128, 0, stream>>>(qb, kbuf, vt, flags, lens, headb);
      else if (g == 1) k_attn<1><<<ga, 128, 0, stream>>>(qb, kbuf, vt, flags, lens, headb);
      else             k_attn<2><<<ga, 128, 0, stream>>>(qb, kbuf, vt, flags, lens, headb);
      k_gemm<4><<<gg, 256, 0, stream>>>(headb, woT + lg, 256, nullptr, cat, x, nullptr, g * Ee);
    }
    k_gemm<5><<<gg, 256, 0, stream>>>(cat, wfT + (size_t)l * 196608, 768, hbuf, nullptr, nullptr, bfp + l * Ee, 0);
    k_bnstats<<<Ee, 256, 0, stream>>>(hbuf, mv);
    k_bnapply<<<Mrows, 256, 0, stream>>>(hbuf, mv, g1 + l * Ee, b1 + l * Ee, hbn, hbn16);
    k_gemm<6><<<gg, 256, 0, stream>>>(hbn16, wffT + (size_t)l * 65536, 256, hbuf, nullptr, hbn, bffp + l * Ee, 0);
    k_bnstats<<<Ee, 256, 0, stream>>>(hbuf, mv);
    k_bnapply<<<Mrows, 256, 0, stream>>>(hbuf, mv, g2 + l * Ee, b2 + l * Ee, x, xh);
  }
  k_final<<<Bb, 256, 0, stream>>>(x, mask, lens, (float*)d_out);
}
